// fc_block_61795989455335
// MI455X (gfx1250) — 8-bit matrix operand order repaired and checked by reading, compiled, not run on hardware
//
#include <hip/hip_runtime.h>

typedef int   v8i __attribute__((ext_vector_type(8)));
typedef float v4f __attribute__((ext_vector_type(4)));

#define B_ROWS   65536
#define FIN      256
#define FOUT     256
#define QMAX_A   15.0f
#define QMAX_W   7.0f
#define PSTEP    1792.0f
#define BN_EPS   1e-5f


__global__ __launch_bounds__(256) void prep_kernel(const float* __restrict__ w,
                                                   float* __restrict__ scale_p,
                                                   int* __restrict__ gsum,
                                                   int* __restrict__ gss) {
    __shared__ float red[256];
    int tid = threadIdx.x;
    float m = 0.f;
    for (int i = tid; i < FOUT * FIN; i += 256) m = fmaxf(m, fabsf(w[i]));
    red[tid] = m;
    __syncthreads();
    for (int s = 128; s > 0; s >>= 1) {
        if (tid < s) red[tid] = fmaxf(red[tid], red[tid + s]);
        __syncthreads();
    }
    if (tid == 0) scale_p[0] = red[0] / QMAX_W;
    gsum[tid] = 0;
    gss[tid]  = 0;
}

__global__ __launch_bounds__(256) void pack_w_kernel(const float* __restrict__ w,
                                                     const float* __restrict__ scale_p,
                                                     unsigned* __restrict__ wsB) {
    int t    = blockIdx.x * 256 + threadIdx.x;
    int lane = t & 31;
    int frag = t >> 5;
    int nblk = frag & 15;
    int c    = frag >> 4;
    int hi   = lane >> 4;
    int n    = nblk * 16 + (lane & 15);
    float inv_s = 1.0f / scale_p[0];
    const float* wr = w + n * FIN;
#pragma unroll
    for (int v = 0; v < 8; ++v) {
        int kk = c * 64 + ((v >> 2) << 5) + (hi << 4) + ((v & 3) << 2);
        unsigned d = 0;
#pragma unroll
        for (int b = 0; b < 4; ++b) {
            float f = fminf(fmaxf(wr[kk + b] * inv_s, -QMAX_W), QMAX_W);
            int q = (int)rintf(f);
            d |= ((unsigned)(q & 0xFF)) << (8 * b);
        }
        wsB[(frag * 32 + lane) * 8 + v] = d;
    }
}

__global__ __launch_bounds__(256) void gemm_kernel(const float* __restrict__ x,
                                                   const unsigned* __restrict__ wsB,
                                                   signed char* __restrict__ ksum,
                                                   int* __restrict__ gsum,
                                                   int* __restrict__ gss) {
    __shared__ int lsum[256];
    __shared__ int lss[256];
    int tid = threadIdx.x;
    lsum[tid] = 0;
    lss[tid]  = 0;
    __syncthreads();

    int wave   = tid >> 5;
    int lane   = tid & 31;
    int rowblk = blockIdx.x * 8 + wave;
    int m0     = rowblk * 16;
    int row    = lane & 15;
    int hi     = lane >> 4;

    v8i A[4];
    const float* xr = x + (size_t)(m0 + row) * FIN;
#pragma unroll
    for (int c = 0; c < 4; ++c) {
#pragma unroll
        for (int a = 0; a < 8; ++a) {
            int kk = c * 64 + ((a >> 2) << 5) + (hi << 4) + ((a & 3) << 2);
            v4f f = __builtin_nontemporal_load((const v4f*)(xr + kk));
            int q0 = (int)rintf(fminf(fmaxf(f.x, 0.f), 1.f) * QMAX_A);
            int q1 = (int)rintf(fminf(fmaxf(f.y, 0.f), 1.f) * QMAX_A);
            int q2 = (int)rintf(fminf(fmaxf(f.z, 0.f), 1.f) * QMAX_A);
            int q3 = (int)rintf(fminf(fmaxf(f.w, 0.f), 1.f) * QMAX_A);
            A[c][a] = q0 | (q1 << 8) | (q2 << 16) | (q3 << 24);
        }
    }

    const float inv_step = 1.0f / PSTEP;

#pragma unroll 2
    for (int nblk = 0; nblk < 16; ++nblk) {
        v8i b0 = *(const v8i*)(wsB + (((0 * 16 + nblk) * 32) + lane) * 8);
        v8i b1 = *(const v8i*)(wsB + (((1 * 16 + nblk) * 32) + lane) * 8);
        v8i b2 = *(const v8i*)(wsB + (((2 * 16 + nblk) * 32) + lane) * 8);
        v8i b3 = *(const v8i*)(wsB + (((3 * 16 + nblk) * 32) + lane) * 8);

        v8i acc0 = {0, 0, 0, 0, 0, 0, 0, 0};
        acc0 = __builtin_amdgcn_wmma_i32_16x16x64_iu8(false, A[0], true, b0, acc0, false, false);
        acc0 = __builtin_amdgcn_wmma_i32_16x16x64_iu8(false, A[1], true, b1, acc0, false, false);
        v8i acc1 = {0, 0, 0, 0, 0, 0, 0, 0};
        acc1 = __builtin_amdgcn_wmma_i32_16x16x64_iu8(false, A[2], true, b2, acc1, false, false);
        acc1 = __builtin_amdgcn_wmma_i32_16x16x64_iu8(false, A[3], true, b3, acc1, false, false);

        int n = nblk * 16 + row;
        int s = 0, ss = 0;
#pragma unroll
        for (int i = 0; i < 8; ++i) {
            int k0 = (int)rintf((float)acc0[i] * inv_step);
            int k1 = (int)rintf((float)acc1[i] * inv_step);
            int k  = k0 + k1;
            s  += k;
            ss += k * k;
            ksum[(size_t)(m0 + i + hi * 8) * FOUT + n] = (signed char)k;
        }
        atomicAdd(&lsum[n], s);
        atomicAdd(&lss[n], ss);
    }

    __syncthreads();
    atomicAdd(&gsum[tid], lsum[tid]);
    atomicAdd(&gss[tid],  lss[tid]);
}

__global__ __launch_bounds__(256) void finalize_kernel(const int* __restrict__ gsum,
                                                       const int* __restrict__ gss,
                                                       const float* __restrict__ scale_p,
                                                       const float* __restrict__ gamma,
                                                       const float* __restrict__ beta,
                                                       float* __restrict__ An,
                                                       float* __restrict__ Bn) {
    int n = threadIdx.x;
    float cc   = PSTEP * scale_p[0] * (1.0f / QMAX_A);
    float invB = 1.0f / (float)B_ROWS;
    float mean = (float)gsum[n] * cc * invB;
    float ex2  = (float)gss[n] * cc * cc * invB;
    float var  = ex2 - mean * mean;
    float rstd = rsqrtf(var + BN_EPS);
    float g    = gamma[n];
    An[n] = cc * rstd * g;
    Bn[n] = beta[n] - mean * rstd * g;
}

__device__ __forceinline__ float dq(int k, float a, float b) {
    float y = fmaf((float)k, a, b);
    y = fminf(fmaxf(y, 0.f), 1.f);
    return rintf(y * QMAX_A) * (1.0f / QMAX_A);
}

__global__ __launch_bounds__(256) void out_kernel(const signed char* __restrict__ ksum,
                                                  const float* __restrict__ An,
                                                  const float* __restrict__ Bn,
                                                  float* __restrict__ out) {
    __shared__ float sA[256];
    __shared__ float sB[256];
    int tid = threadIdx.x;
    sA[tid] = An[tid];
    sB[tid] = Bn[tid];
    __syncthreads();

    size_t idx4 = (size_t)blockIdx.x * 256 + tid;
    size_t base = idx4 * 4;
    char4 k4 = ((const char4*)ksum)[idx4];
    int n0 = (int)(base & (FOUT - 1));
    v4f o;
    o.x = dq(k4.x, sA[n0 + 0], sB[n0 + 0]);
    o.y = dq(k4.y, sA[n0 + 1], sB[n0 + 1]);
    o.z = dq(k4.z, sA[n0 + 2], sB[n0 + 2]);
    o.w = dq(k4.w, sA[n0 + 3], sB[n0 + 3]);
    __builtin_nontemporal_store(o, (v4f*)out + idx4);
}

extern "C" void kernel_launch(void* const* d_in, const int* in_sizes, int n_in,
                              void* d_out, int out_size, void* d_ws, size_t ws_size,
                              hipStream_t stream) {
    const float* x     = (const float*)d_in[0];
    const float* w     = (const float*)d_in[1];
    const float* gamma = (const float*)d_in[2];
    const float* beta  = (const float*)d_in[3];
    float* out = (float*)d_out;

    char* ws = (char*)d_ws;
    unsigned*    wsB     = (unsigned*)(ws + 0);
    float*       scale_p = (float*)(ws + 65536);
    int*         gsum    = (int*)(ws + 66560);
    int*         gss     = (int*)(ws + 67584);
    float*       An      = (float*)(ws + 68608);
    float*       Bn      = (float*)(ws + 69632);
    signed char* ksum    = (signed char*)(ws + 131072);

    prep_kernel<<<1, 256, 0, stream>>>(w, scale_p, gsum, gss);
    pack_w_kernel<<<8, 256, 0, stream>>>(w, scale_p, wsB);
    gemm_kernel<<<B_ROWS / 16 / 8, 256, 0, stream>>>(x, wsB, ksum, gsum, gss);
    finalize_kernel<<<1, 256, 0, stream>>>(gsum, gss, scale_p, gamma, beta, An, Bn);
    out_kernel<<<(B_ROWS * FOUT / 4) / 256, 256, 0, stream>>>(ksum, An, Bn, out);
}
